// MultiHeadAttention_42107859370754
// MI455X (gfx1250) — matrix operand order repaired and checked by reading, not compiled, not run on hardware
//
#include <hip/hip_runtime.h>
#include <hip/hip_bf16.h>

typedef __attribute__((ext_vector_type(16))) _Float16 v16h;
typedef __attribute__((ext_vector_type(8)))  _Float16 v8h;
typedef __attribute__((ext_vector_type(4)))  _Float16 v4h;
typedef __attribute__((ext_vector_type(8)))  float    v8f;
typedef __attribute__((ext_vector_type(4)))  float    v4f;

#define B_   2
#define S_   2048
#define H_   1024
#define NH_  16
#define HD_  64
#define LDK  40   // padded K-stride (halves) in LDS, keeps 16B alignment (80B rows)

static __device__ __forceinline__ v16h cat8(v8h a, v8h b) {
    return __builtin_shufflevector(a, b, 0,1,2,3,4,5,6,7,8,9,10,11,12,13,14,15);
}
static __device__ __forceinline__ v8f wmma_f16(v16h a, v16h b, v8f c) {
    return __builtin_amdgcn_wmma_f32_16x16x32_f16(false, a, false, b, (short)0, c, false, false);
}
// Low 32 bits of a generic pointer to LDS == byte offset within the
// workgroup's LDS allocation (addrspacecast puts the aperture in the high half).
static __device__ __forceinline__ uint32_t lds_addr32(const void* p) {
    return (uint32_t)(uintptr_t)p;
}
// CDNA5 async global->LDS copy, 16B per lane, tracked by ASYNCcnt.
static __device__ __forceinline__ void async_copy_b128(uint32_t lds_off, const void* gptr) {
    asm volatile("global_load_async_to_lds_b128 %0, %1, off"
                 :: "v"(lds_off), "v"((uint64_t)(uintptr_t)gptr)
                 : "memory");
}
static __device__ __forceinline__ void wait_async0() {
    asm volatile("s_wait_asynccnt 0x0" ::: "memory");
}

// ---------------------------------------------------------------------------
// Kernel 1: fused QKV projection.  C[4096x1024] = x[4096x1024] * W + bias
//   z=0 -> q (scaled by 1/8)  laid out [B,NH,S,HD]  (f16)
//   z=1 -> k                  laid out [B,NH,S,HD]  (f16)
//   z=2 -> v                  laid out [B,NH,HD,S]  (f16, transposed for P*V)
// grid (32, 8, 3), block 256 (8 waves). 128x128 output tile per block.
// ---------------------------------------------------------------------------
__global__ __launch_bounds__(256)
void qkv_proj_kernel(const float* __restrict__ x,
                     const float* __restrict__ Wq, const float* __restrict__ bq,
                     const float* __restrict__ Wk, const float* __restrict__ bk,
                     const float* __restrict__ Wv, const float* __restrict__ bv,
                     _Float16* __restrict__ qf, _Float16* __restrict__ kf,
                     _Float16* __restrict__ vT)
{
    __shared__ __align__(16) _Float16 lx[128 * LDK];   // x tile   [row][k]
    __shared__ __align__(16) _Float16 lw[128 * LDK];   // W tile T [n][k]

    const int z  = blockIdx.z;
    const float* Wm = (z == 0) ? Wq : (z == 1) ? Wk : Wv;
    const float* bm = (z == 0) ? bq : (z == 1) ? bk : bv;

    const int m0 = blockIdx.x * 128;
    const int n0 = blockIdx.y * 128;
    const int tid  = threadIdx.x;
    const int wave = tid >> 5;
    const int lane = tid & 31;
    const int l16  = lane & 15;
    const int hi   = lane >> 4;

    v8f acc[8];
    #pragma unroll
    for (int ns = 0; ns < 8; ++ns) acc[ns] = 0.0f;

    for (int kk = 0; kk < H_; kk += 32) {
        // ---- stage x chunk [128 x 32] fp32 -> f16 LDS ----
        const float* xg = x + (size_t)m0 * H_ + kk;
        #pragma unroll
        for (int i = 0; i < 4; ++i) {
            int idx = tid + i * 256;            // 1024 float4s
            int row = idx >> 3;
            int c4  = (idx & 7) << 2;
            v4f xv = *(const v4f*)(xg + (size_t)row * H_ + c4);
            v4h hv; hv.x = (_Float16)xv.x; hv.y = (_Float16)xv.y;
            hv.z = (_Float16)xv.z; hv.w = (_Float16)xv.w;
            *(v4h*)&lx[row * LDK + c4] = hv;
        }
        // ---- stage W chunk [32 x 128] fp32 -> f16 LDS, transposed to [n][k] ----
        const float* wg = Wm + (size_t)kk * H_ + n0;
        #pragma unroll
        for (int i = 0; i < 4; ++i) {
            int idx  = tid + i * 256;
            int krow = idx >> 5;
            int c4   = (idx & 31) << 2;
            v4f wv = *(const v4f*)(wg + (size_t)krow * H_ + c4);
            lw[(c4 + 0) * LDK + krow] = (_Float16)wv.x;
            lw[(c4 + 1) * LDK + krow] = (_Float16)wv.y;
            lw[(c4 + 2) * LDK + krow] = (_Float16)wv.z;
            lw[(c4 + 3) * LDK + krow] = (_Float16)wv.w;
        }
        __syncthreads();

        // ---- A fragment: 16x32 f16, lane holds row = wave*16 + l16 ----
        const int arow = wave * 16 + l16;
        v8h a0 = *(const v8h*)&lx[arow * LDK + hi * 8];
        v8h a1 = *(const v8h*)&lx[arow * LDK + 16 + hi * 8];
        v16h A = cat8(a0, a1);

        #pragma unroll
        for (int ns = 0; ns < 8; ++ns) {
            const int n = ns * 16 + l16;
            v8h b0 = *(const v8h*)&lw[n * LDK + hi * 8];
            v8h b1 = *(const v8h*)&lw[n * LDK + 16 + hi * 8];
            acc[ns] = wmma_f16(A, cat8(b0, b1), acc[ns]);
        }
        __syncthreads();
    }

    // ---- bias + scale + scatter f16 to q/k/v layouts ----
    const float scale = (z == 0) ? 0.125f : 1.0f;   // 1/sqrt(64) folded into q
    #pragma unroll
    for (int ns = 0; ns < 8; ++ns) {
        const int n_g = n0 + ns * 16 + l16;
        const float bvv = bm[n_g];
        const int nh = n_g >> 6;
        const int d  = n_g & 63;
        #pragma unroll
        for (int r = 0; r < 8; ++r) {
            const int row_g = m0 + wave * 16 + r + 8 * hi;
            const int b = row_g >> 11;          // / S_
            const int s = row_g & 2047;
            _Float16 h = (_Float16)((acc[ns][r] + bvv) * scale);
            if (z == 2)
                vT[(((size_t)b * NH_ + nh) * HD_ + d) * S_ + s] = h;
            else if (z == 1)
                kf[(((size_t)b * NH_ + nh) * S_ + s) * HD_ + d] = h;
            else
                qf[(((size_t)b * NH_ + nh) * S_ + s) * HD_ + d] = h;
        }
    }
}

// ---------------------------------------------------------------------------
// Kernel 2: flash-style attention.  grid (B*NH, S/128), block 256 (8 waves).
// K/V chunks are staged into double-buffered LDS with CDNA5 async copies
// (global_load_async_to_lds_b128 / ASYNCcnt) so the next chunk's copy
// overlaps this chunk's WMMA + softmax. P transposed via per-wave LDS.
// ---------------------------------------------------------------------------
__global__ __launch_bounds__(256)
void attn_kernel(const _Float16* __restrict__ qf, const _Float16* __restrict__ kf,
                 const _Float16* __restrict__ vT, const int* __restrict__ mask,
                 float* __restrict__ out)
{
    __shared__ __align__(16) _Float16 lk[2][128 * HD_];   // K chunk  [j][d]   (2x16KB)
    __shared__ __align__(16) _Float16 lv[2][HD_ * 128];   // V chunk T [d][j]  (2x16KB)
    __shared__ __align__(16) _Float16 lp[8 * 16 * 128];   // per-wave P tiles  (32KB)

    const int bh = blockIdx.x;            // b*NH + nh
    const int b  = bh >> 4;
    const int nh = bh & 15;
    const int q0 = blockIdx.y * 128;
    const int tid  = threadIdx.x;
    const int wave = tid >> 5;
    const int lane = tid & 31;
    const int l16  = lane & 15;
    const int hi   = lane >> 4;

    const _Float16* qh = qf + (size_t)bh * S_ * HD_;
    const _Float16* kh = kf + (size_t)bh * S_ * HD_;
    const _Float16* vh = vT + (size_t)bh * HD_ * S_;
    const int* mrow = mask + ((size_t)b * S_ + q0 + wave * 16) * S_;
    _Float16* Pw = lp + wave * (16 * 128);

    // Issue the async stage of one 128-key chunk (K: 16KB contiguous; V: 64
    // rows of 256B). 8 x b128 per thread -> ASYNCcnt tracks 8 per wave.
    auto stage_chunk = [&](int jb, int buf) {
        const char* ksrc = (const char*)(kh + (size_t)jb * HD_);
        const uint32_t lkoff = lds_addr32(&lk[buf][0]);
        const uint32_t lvoff = lds_addr32(&lv[buf][0]);
        #pragma unroll
        for (int i = 0; i < 4; ++i) {
            int idx = tid + i * 256;                  // 0..1023 x 16B
            async_copy_b128(lkoff + idx * 16, ksrc + (size_t)idx * 16);
        }
        #pragma unroll
        for (int i = 0; i < 4; ++i) {
            int idx = tid + i * 256;
            int d = idx >> 4, seg = idx & 15;
            async_copy_b128(lvoff + (d * 128 + seg * 8) * 2,
                            vh + (size_t)d * S_ + jb + seg * 8);
        }
    };

    // Q fragments (resident): two K-steps over d=64
    const int qrow = q0 + wave * 16 + l16;
    v16h Aq[2];
    #pragma unroll
    for (int ks = 0; ks < 2; ++ks) {
        const _Float16* qp = qh + (size_t)qrow * HD_ + ks * 32;
        v8h a0 = *(const v8h*)(qp + hi * 8);
        v8h a1 = *(const v8h*)(qp + 16 + hi * 8);
        Aq[ks] = cat8(a0, a1);
    }

    v8f o[4];
    #pragma unroll
    for (int ns = 0; ns < 4; ++ns) o[ns] = 0.0f;
    float m[8], l[8];
    #pragma unroll
    for (int r = 0; r < 8; ++r) { m[r] = -3.0e38f; l[r] = 0.0f; }

    stage_chunk(0, 0);

    for (int ci = 0; ci < S_ / 128; ++ci) {
        const int jb = ci * 128;
        const _Float16* lkb = &lk[ci & 1][0];
        const _Float16* lvb = &lv[ci & 1][0];

        wait_async0();        // own chunk-ci copies landed
        __syncthreads();      // everyone's copies landed; other buffer free
        if (ci + 1 < S_ / 128) stage_chunk(jb + 128, (ci + 1) & 1);

        // ---- scores S = (q/sqrt(d)) K^T  (q pre-scaled) ----
        v8f sc[8];
        #pragma unroll
        for (int ns = 0; ns < 8; ++ns) sc[ns] = 0.0f;
        #pragma unroll
        for (int ns = 0; ns < 8; ++ns) {
            const int jl = ns * 16 + l16;
            #pragma unroll
            for (int ks = 0; ks < 2; ++ks) {
                const _Float16* kp = lkb + jl * HD_ + ks * 32;
                v8h b0 = *(const v8h*)(kp + hi * 8);
                v8h b1 = *(const v8h*)(kp + 16 + hi * 8);
                sc[ns] = wmma_f16(Aq[ks], cat8(b0, b1), sc[ns]);
            }
        }

        // ---- additive mask (streaming: non-temporal) + row max ----
        float rmax[8];
        #pragma unroll
        for (int r = 0; r < 8; ++r) rmax[r] = -3.0e38f;
        #pragma unroll
        for (int ns = 0; ns < 8; ++ns) {
            const int j = jb + ns * 16 + l16;
            #pragma unroll
            for (int r = 0; r < 8; ++r) {
                const int rloc = r + 8 * hi;
                int mv = __builtin_nontemporal_load(mrow + (size_t)rloc * S_ + j);
                float sv = sc[ns][r] - 10000.0f * (float)mv;
                sc[ns][r] = sv;
                rmax[r] = fmaxf(rmax[r], sv);
            }
        }
        #pragma unroll
        for (int r = 0; r < 8; ++r) {
            #pragma unroll
            for (int off = 1; off < 16; off <<= 1)
                rmax[r] = fmaxf(rmax[r], __shfl_xor(rmax[r], off, 32));
        }

        // ---- online softmax rescale ----
        float ps[8];
        #pragma unroll
        for (int r = 0; r < 8; ++r) {
            float mn = fmaxf(m[r], rmax[r]);
            float alpha = __expf(m[r] - mn);
            m[r] = mn;
            l[r] *= alpha;
            ps[r] = 0.0f;
            #pragma unroll
            for (int ns = 0; ns < 4; ++ns) o[ns][r] *= alpha;
        }

        // ---- P = exp(S - m): row sums + stage transposed in LDS ----
        #pragma unroll
        for (int ns = 0; ns < 8; ++ns) {
            #pragma unroll
            for (int r = 0; r < 8; ++r) {
                float p = __expf(sc[ns][r] - m[r]);
                ps[r] += p;
                Pw[(r + 8 * hi) * 128 + ns * 16 + l16] = (_Float16)p;
            }
        }
        #pragma unroll
        for (int r = 0; r < 8; ++r) {
            #pragma unroll
            for (int off = 1; off < 16; off <<= 1)
                ps[r] += __shfl_xor(ps[r], off, 32);
            l[r] += ps[r];
        }
        // order the per-wave LDS P stores before the A-fragment reads
        asm volatile("s_wait_dscnt 0x0" ::: "memory");

        // ---- O += P * V  (V chunk transposed in LDS: lvb[d][j]) ----
        #pragma unroll
        for (int ks2 = 0; ks2 < 4; ++ks2) {
            const _Float16* pp = Pw + l16 * 128 + ks2 * 32;
            v8h a0 = *(const v8h*)(pp + hi * 8);
            v8h a1 = *(const v8h*)(pp + 16 + hi * 8);
            v16h Ap = cat8(a0, a1);
            #pragma unroll
            for (int ns = 0; ns < 4; ++ns) {
                const int d = ns * 16 + l16;
                const _Float16* vp = lvb + d * 128 + ks2 * 32;
                v8h b0 = *(const v8h*)(vp + hi * 8);
                v8h b1 = *(const v8h*)(vp + 16 + hi * 8);
                o[ns] = wmma_f16(Ap, cat8(b0, b1), o[ns]);
            }
        }
        __syncthreads();      // all waves done reading lk/lv[ci&1] before it is refilled
    }

    // ---- normalize by l and write out[b][s][nh*64+d] (fp32) ----
    float rl[8];
    #pragma unroll
    for (int r = 0; r < 8; ++r) rl[r] = 1.0f / l[r];
    float* og = out + ((size_t)b * S_ + q0 + wave * 16) * H_ + nh * HD_;
    #pragma unroll
    for (int ns = 0; ns < 4; ++ns) {
        #pragma unroll
        for (int r = 0; r < 8; ++r) {
            og[(size_t)(r + 8 * hi) * H_ + ns * 16 + l16] = o[ns][r] * rl[r];
        }
    }
}

extern "C" void kernel_launch(void* const* d_in, const int* in_sizes, int n_in,
                              void* d_out, int out_size, void* d_ws, size_t ws_size,
                              hipStream_t stream) {
    (void)in_sizes; (void)n_in; (void)out_size; (void)ws_size;
    const float* x    = (const float*)d_in[0];
    const int*   msk  = (const int*)d_in[1];
    const float* Wq   = (const float*)d_in[2];
    const float* bq   = (const float*)d_in[3];
    const float* Wk   = (const float*)d_in[4];
    const float* bk   = (const float*)d_in[5];
    const float* Wv   = (const float*)d_in[6];
    const float* bv   = (const float*)d_in[7];
    float* out = (float*)d_out;

    const size_t per = (size_t)B_ * NH_ * S_ * HD_;   // 4M halves each
    _Float16* qf = (_Float16*)d_ws;
    _Float16* kf = qf + per;
    _Float16* vT = kf + per;

    dim3 g1(32, 8, 3);     // (M/128, N/128, q|k|v)
    qkv_proj_kernel<<<g1, 256, 0, stream>>>(x, Wq, bq, Wk, bk, Wv, bv, qf, kf, vT);

    dim3 g2(B_ * NH_, S_ / 128);
    attn_kernel<<<g2, 256, 0, stream>>>(qf, kf, vT, msk, out);
}
